// DeformableDepthLSSTransform_35330400977102
// MI455X (gfx1250) — matrix operand order repaired and checked by reading, not compiled, not run on hardware
//
#include <hip/hip_runtime.h>
#include <hip/hip_bf16.h>

// ---------------------------------------------------------------------------
// Types for CDNA5 WMMA (wave32, v_wmma_f32_16x16x32_bf16)
// ---------------------------------------------------------------------------
typedef __bf16 bhalf_t;
typedef bhalf_t bhalf8  __attribute__((ext_vector_type(8)));
typedef bhalf_t bhalf16 __attribute__((ext_vector_type(16)));
typedef float   floatx8 __attribute__((ext_vector_type(8)));

#define DEVFN static __device__ __forceinline__

DEVFN bhalf_t f2bf(float f) {
    unsigned u = __float_as_uint(f);
    u += 0x7FFFu + ((u >> 16) & 1u);          // round-to-nearest-even
    unsigned short h = (unsigned short)(u >> 16);
    return __builtin_bit_cast(bhalf_t, h);
}

// ---------------------------------------------------------------------------
// Problem constants
// ---------------------------------------------------------------------------
#define MIMG   6
#define KNB    8
#define IHH    256
#define IWW    704
#define FHH    32
#define FWW    88
#define NPIX   (FHH*FWW)      /* 2816 */
#define INCH   256
#define OUTC   80
#define DBINS  59
#define ZCH    (DBINS+OUTC)   /* 139 */
#define NXV    360
#define NYV    360
#define BEVHW  (NXV*NYV)      /* 129600 */

// ---------------------------------------------------------------------------
// Weight prep: fold BN gamma into weights, transpose to [K][Cout] bf16,
// produce fused per-channel bias (b*g + beta).
// ---------------------------------------------------------------------------
__global__ void prep_weights_kernel(const float* __restrict__ w,
                                    const float* __restrict__ b,
                                    const float* __restrict__ g,
                                    const float* __restrict__ beta,
                                    bhalf_t* __restrict__ wt,
                                    float* __restrict__ bout,
                                    int Cout, int K) {
    int i = blockIdx.x * blockDim.x + threadIdx.x;
    if (i < K * Cout) {
        int k = i / Cout, co = i - k * Cout;
        float s = g ? g[co] : 1.0f;
        wt[i] = f2bf(w[co * K + k] * s);
    }
    if (i < Cout) {
        float s  = g ? g[i] : 1.0f;
        float bb = (b ? b[i] * s : 0.0f) + (beta ? beta[i] : 0.0f);
        bout[i] = bb;
    }
}

// ---------------------------------------------------------------------------
// Stage 1: depth attention fuse + attn 1x1 conv + d1 (1->8 1x1) + BN + ReLU
// out: h1 [6][8][256][704]
// ---------------------------------------------------------------------------
__global__ void attn_d1_kernel(const float* __restrict__ depth,
                               const float* __restrict__ nbd,
                               const float* __restrict__ attn_w,
                               const float* __restrict__ attn_b,
                               const float* __restrict__ d1w,
                               const float* __restrict__ d1b,
                               const float* __restrict__ g1,
                               const float* __restrict__ b1,
                               float* __restrict__ h1) {
    int i = blockIdx.x * blockDim.x + threadIdx.x;
    const int HW = IHH * IWW;
    if (i >= MIMG * HW) return;
    int m = i / HW, pix = i - m * HW;
    float d = depth[i];
    float v[KNB], a[KNB];
    float mx = -1e30f;
    #pragma unroll
    for (int k = 0; k < KNB; ++k) {
        v[k] = nbd[((size_t)(m * KNB + k)) * HW + pix];
        a[k] = -fabsf(d - v[k]);
        mx = fmaxf(mx, a[k]);
    }
    float se = 0.f, acc = 0.f;
    #pragma unroll
    for (int k = 0; k < KNB; ++k) {
        float e = __expf(a[k] - mx);
        se += e; acc += e * v[k];
    }
    float fused = acc / se;
    fused = fused * attn_w[0] + attn_b[0];
    #pragma unroll
    for (int c = 0; c < 8; ++c) {
        float t = (fused * d1w[c] + d1b[c]) * g1[c] + b1[c];
        h1[((size_t)(m * 8 + c)) * HW + pix] = fmaxf(t, 0.f);
    }
}

// ---------------------------------------------------------------------------
// d2: 8->32 5x5 stride4 pad2, [256,704]->[64,176], BN+ReLU  -> h2
// ---------------------------------------------------------------------------
__global__ void d2_kernel(const float* __restrict__ in, const float* __restrict__ w,
                          const float* __restrict__ b, const float* __restrict__ g,
                          const float* __restrict__ beta, float* __restrict__ out) {
    const int OH = 64, OW = 176, H = IHH, W = IWW, CI = 8, CO = 32;
    int i = blockIdx.x * blockDim.x + threadIdx.x;
    int total = MIMG * CO * OH * OW;
    if (i >= total) return;
    int m = i / (CO * OH * OW); int r = i - m * (CO * OH * OW);
    int co = r / (OH * OW); r -= co * (OH * OW);
    int oy = r / OW, ox = r - oy * OW;
    float acc = 0.f;
    for (int ci = 0; ci < CI; ++ci) {
        const float* ip = in + ((size_t)(m * CI + ci)) * H * W;
        const float* wp = w + ((co * CI + ci) * 25);
        #pragma unroll
        for (int ky = 0; ky < 5; ++ky) {
            int iy = oy * 4 + ky - 2;
            if (iy < 0 || iy >= H) continue;
            #pragma unroll
            for (int kx = 0; kx < 5; ++kx) {
                int ix = ox * 4 + kx - 2;
                if (ix < 0 || ix >= W) continue;
                acc += ip[iy * W + ix] * wp[ky * 5 + kx];
            }
        }
    }
    float t = (acc + b[co]) * g[co] + beta[co];
    out[i] = fmaxf(t, 0.f);
}

// ---------------------------------------------------------------------------
// d3: 32->64 5x5 stride2 pad2, [64,176]->[32,88], BN+ReLU, write into
// z concat buffer channels [0..63]  (z: [6][320][2816])
// ---------------------------------------------------------------------------
__global__ void d3_kernel(const float* __restrict__ in, const float* __restrict__ w,
                          const float* __restrict__ b, const float* __restrict__ g,
                          const float* __restrict__ beta, float* __restrict__ z) {
    const int H = 64, W = 176, CI = 32, CO = 64;
    int i = blockIdx.x * blockDim.x + threadIdx.x;
    int total = MIMG * CO * NPIX;
    if (i >= total) return;
    int m = i / (CO * NPIX); int r = i - m * (CO * NPIX);
    int co = r / NPIX; r -= co * NPIX;
    int oy = r / FWW, ox = r - oy * FWW;
    float acc = 0.f;
    for (int ci = 0; ci < CI; ++ci) {
        const float* ip = in + ((size_t)(m * CI + ci)) * H * W;
        const float* wp = w + ((co * CI + ci) * 25);
        #pragma unroll
        for (int ky = 0; ky < 5; ++ky) {
            int iy = oy * 2 + ky - 2;
            if (iy < 0 || iy >= H) continue;
            #pragma unroll
            for (int kx = 0; kx < 5; ++kx) {
                int ix = ox * 2 + kx - 2;
                if (ix < 0 || ix >= W) continue;
                acc += ip[iy * W + ix] * wp[ky * 5 + kx];
            }
        }
    }
    float t = (acc + b[co]) * g[co] + beta[co];
    z[((size_t)(m * 320 + co)) * NPIX + (oy * FWW + ox)] = fmaxf(t, 0.f);
}

// ---------------------------------------------------------------------------
// Copy img [6][256][2816] into z channels 64..319
// ---------------------------------------------------------------------------
__global__ void img_concat_kernel(const float* __restrict__ img, float* __restrict__ z) {
    int i = blockIdx.x * blockDim.x + threadIdx.x;
    int total = MIMG * INCH * NPIX;
    if (i >= total) return;
    int m = i / (INCH * NPIX); int r = i - m * (INCH * NPIX);
    int c = r / NPIX; int p = r - c * NPIX;
    z[((size_t)(m * 320 + 64 + c)) * NPIX + p] = img[i];
}

// ---------------------------------------------------------------------------
// Generic implicit-GEMM conv via bf16 WMMA (f32 accum), templated on the
// static conv geometry so the im2col index math folds to constant-magic ops.
// in  : [M][Cin][H][W] f32
// wt  : [K=Cin*KS*KS][Cout] bf16 (BN-gamma folded)
// out : [M][Cout][OH][OWt] f32;  bias+ReLU epilogue
// Block: 256 thr (8 waves). Tile: 64 out-ch x 128 pixels.
//   wave w: co off = (w&1)*32, pix off = (w>>1)*32 -> 2x2 sub-tiles,
//   4 WMMAs per K-chunk with A/B fragment reuse.
// LDS column-major ([col][k], 16B-aligned rows) -> contiguous b128 loads.
// ---------------------------------------------------------------------------
template <int KS, int STRIDE, int PAD, int RELU, int OWt>
__global__ void __launch_bounds__(256)
conv_wmma_kernel(const float* __restrict__ in, const bhalf_t* __restrict__ wt,
                 const float* __restrict__ bias, float* __restrict__ out,
                 int Cin, int Cout, int H, int W, int OH) {
    __shared__ __align__(16) bhalf_t ldsW[64][40];
    __shared__ __align__(16) bhalf_t ldsP[128][40];

    constexpr int KSQ = KS * KS;
    const int tid  = threadIdx.x;
    const int wave = tid >> 5, lane = tid & 31;
    const int m       = blockIdx.z;
    const int pixBase = blockIdx.x * 128;
    const int coBase  = blockIdx.y * 64;
    const int npix = OH * OWt;
    const int Ktot = Cin * KSQ;

    const int waveCo  = (wave & 1) * 32;
    const int wavePix = (wave >> 1) * 32;
    const int lm  = lane & 15;
    const int hiA = (lane >= 16) ? 16 : 0;   // A K-offset (contiguous k, matches B)
    const int hiB = (lane >= 16) ? 16 : 0;   // B K-offset (contiguous k, matches A)
    const bhalf_t BZ = __builtin_bit_cast(bhalf_t, (unsigned short)0);

    floatx8 acc00 = {}, acc01 = {}, acc10 = {}, acc11 = {};

    for (int kb = 0; kb < Ktot; kb += 32) {
        // prefetch next weight K-chunk (global_prefetch_b8)
        if (kb + 32 < Ktot)
            __builtin_prefetch(wt + (size_t)(kb + 32) * Cout + coBase + (tid & 63), 0, 1);

        // ---- stage weights: 32 K-rows x 64 out-ch -> ldsW[co][k] ----
        #pragma unroll
        for (int j = 0; j < 8; ++j) {
            int lin = tid + j * 256;
            int c = lin & 63, r = lin >> 6;
            int k = kb + r, co = coBase + c;
            ldsW[c][r] = (k < Ktot && co < Cout) ? wt[k * Cout + co] : BZ;
        }
        // ---- stage im2col patches: 32 K-rows x 128 pixels -> ldsP[pix][k] ----
        #pragma unroll
        for (int j = 0; j < 16; ++j) {
            int lin = tid + j * 256;
            int c = lin & 127, r = lin >> 7;
            int k = kb + r;
            int p = pixBase + c;
            float v = 0.f;
            if (k < Ktot && p < npix) {
                int ci  = k / KSQ;
                int rem = k - ci * KSQ;
                int ky  = rem / KS, kx = rem - ky * KS;
                int py  = p / OWt,  px = p - py * OWt;
                int iy  = py * STRIDE + ky - PAD;
                int ix  = px * STRIDE + kx - PAD;
                if (iy >= 0 && iy < H && ix >= 0 && ix < W)
                    v = in[((size_t)(m * Cin + ci) * H + iy) * W + ix];
            }
            ldsP[c][r] = f2bf(v);
        }
        __syncthreads();

        // ---- fragment gathers (contiguous -> ds_load_b128) ----
        bhalf8 b0l = *(const bhalf8*)&ldsP[wavePix + lm][hiB];
        bhalf8 b0h = *(const bhalf8*)&ldsP[wavePix + lm][hiB + 8];
        bhalf16 bm0 = __builtin_shufflevector(b0l, b0h,
            0,1,2,3,4,5,6,7,8,9,10,11,12,13,14,15);
        bhalf8 b1l = *(const bhalf8*)&ldsP[wavePix + 16 + lm][hiB];
        bhalf8 b1h = *(const bhalf8*)&ldsP[wavePix + 16 + lm][hiB + 8];
        bhalf16 bm1 = __builtin_shufflevector(b1l, b1h,
            0,1,2,3,4,5,6,7,8,9,10,11,12,13,14,15);

        bhalf8 a0l = *(const bhalf8*)&ldsW[waveCo + lm][hiA];
        bhalf8 a0h = *(const bhalf8*)&ldsW[waveCo + lm][hiA + 8];
        bhalf16 am0 = __builtin_shufflevector(a0l, a0h,
            0,1,2,3,4,5,6,7,8,9,10,11,12,13,14,15);
        bhalf8 a1l = *(const bhalf8*)&ldsW[waveCo + 16 + lm][hiA];
        bhalf8 a1h = *(const bhalf8*)&ldsW[waveCo + 16 + lm][hiA + 8];
        bhalf16 am1 = __builtin_shufflevector(a1l, a1h,
            0,1,2,3,4,5,6,7,8,9,10,11,12,13,14,15);

        acc00 = __builtin_amdgcn_wmma_f32_16x16x32_bf16(
                    false, am0, false, bm0, (short)0, acc00, false, false);
        acc01 = __builtin_amdgcn_wmma_f32_16x16x32_bf16(
                    false, am0, false, bm1, (short)0, acc01, false, false);
        acc10 = __builtin_amdgcn_wmma_f32_16x16x32_bf16(
                    false, am1, false, bm0, (short)0, acc10, false, false);
        acc11 = __builtin_amdgcn_wmma_f32_16x16x32_bf16(
                    false, am1, false, bm1, (short)0, acc11, false, false);
        __syncthreads();
    }

    // ---- epilogue: bias (+ReLU), scatter per ISA C/D layout ----
    const int n0   = pixBase + wavePix + lm;
    const int n1   = n0 + 16;
    const int mrow = (lane >= 16) ? 8 : 0;
    #pragma unroll
    for (int r = 0; r < 8; ++r) {
        int co0 = coBase + waveCo + r + mrow;
        int co1 = co0 + 16;
        if (co0 < Cout) {
            float bz = bias[co0];
            float* op = out + ((size_t)m * Cout + co0) * npix;
            if (n0 < npix) {
                float v = acc00[r] + bz;
                op[n0] = RELU ? fmaxf(v, 0.f) : v;
            }
            if (n1 < npix) {
                float v = acc01[r] + bz;
                op[n1] = RELU ? fmaxf(v, 0.f) : v;
            }
        }
        if (co1 < Cout) {
            float bz = bias[co1];
            float* op = out + ((size_t)m * Cout + co1) * npix;
            if (n0 < npix) {
                float v = acc10[r] + bz;
                op[n0] = RELU ? fmaxf(v, 0.f) : v;
            }
            if (n1 < npix) {
                float v = acc11[r] + bz;
                op[n1] = RELU ? fmaxf(v, 0.f) : v;
            }
        }
    }
}

// ---------------------------------------------------------------------------
// Zero fill
// ---------------------------------------------------------------------------
__global__ void zero_kernel(float* __restrict__ p, int n) {
    int i = blockIdx.x * blockDim.x + threadIdx.x;
    if (i < n) p[i] = 0.f;
}

// ---------------------------------------------------------------------------
// Depth softmax + frustum geometry + BEV scatter-add.
// One block per (image, pixel). zo: [6][139][2816]. bev: [80][360][360].
// ---------------------------------------------------------------------------
DEVFN void inv3(const float* a, float* o) {
    float a00=a[0],a01=a[1],a02=a[2],a10=a[3],a11=a[4],a12=a[5],a20=a[6],a21=a[7],a22=a[8];
    float det = a00*(a11*a22-a12*a21) - a01*(a10*a22-a12*a20) + a02*(a10*a21-a11*a20);
    float id = 1.0f / det;
    o[0] =  (a11*a22 - a12*a21) * id;
    o[1] = -(a01*a22 - a02*a21) * id;
    o[2] =  (a01*a12 - a02*a11) * id;
    o[3] = -(a10*a22 - a12*a20) * id;
    o[4] =  (a00*a22 - a02*a20) * id;
    o[5] = -(a00*a12 - a02*a10) * id;
    o[6] =  (a10*a21 - a11*a20) * id;
    o[7] = -(a00*a21 - a01*a20) * id;
    o[8] =  (a00*a11 - a01*a10) * id;
}

__global__ void __launch_bounds__(128)
bev_scatter_kernel(const float* __restrict__ zo,
                   const float* __restrict__ c2l_rot, const float* __restrict__ c2l_tr,
                   const float* __restrict__ intr,
                   const float* __restrict__ post_rot, const float* __restrict__ post_tr,
                   float* __restrict__ bev) {
    __shared__ float sl[DBINS];
    __shared__ float sf[OUTC];
    __shared__ float sw[DBINS];
    __shared__ int   sc[DBINS];
    __shared__ float sip[9], scb[9], spt[3], sct[3];

    const int bid = blockIdx.x;
    const int m   = bid / NPIX;
    const int pix = bid - m * NPIX;
    const int tid = threadIdx.x;

    if (tid < DBINS) sl[tid] = zo[((size_t)(m * ZCH + tid)) * NPIX + pix];
    if (tid < OUTC)  sf[tid] = zo[((size_t)(m * ZCH + DBINS + tid)) * NPIX + pix];
    if (tid == 0) {
        inv3(post_rot + m * 9, sip);
        float ii[9];
        inv3(intr + m * 9, ii);
        const float* rot = c2l_rot + m * 9;
        for (int i = 0; i < 3; ++i)
            for (int j = 0; j < 3; ++j)
                scb[i*3+j] = rot[i*3+0]*ii[0*3+j] + rot[i*3+1]*ii[1*3+j] + rot[i*3+2]*ii[2*3+j];
        for (int i = 0; i < 3; ++i) { spt[i] = post_tr[m*3+i]; sct[i] = c2l_tr[m*3+i]; }
    }
    __syncthreads();

    if (tid < DBINS) {
        float mx = -1e30f;
        for (int d = 0; d < DBINS; ++d) mx = fmaxf(mx, sl[d]);
        float se = 0.f;
        for (int d = 0; d < DBINS; ++d) se += __expf(sl[d] - mx);
        sw[tid] = __expf(sl[tid] - mx) / se;

        int py = pix / FWW, px = pix - py * FWW;
        float u  = px * (703.0f / 87.0f);       // linspace(0, IW-1, FW)
        float v  = py * (255.0f / 31.0f);       // linspace(0, IH-1, FH)
        float dv = 1.0f + (float)tid;           // arange(1, 60)
        float q0 = u - spt[0], q1 = v - spt[1], q2 = dv - spt[2];
        float r0 = sip[0]*q0 + sip[1]*q1 + sip[2]*q2;
        float r1 = sip[3]*q0 + sip[4]*q1 + sip[5]*q2;
        float r2 = sip[6]*q0 + sip[7]*q1 + sip[8]*q2;
        float s0 = r0 * r2, s1 = r1 * r2, s2 = r2;
        float t0 = scb[0]*s0 + scb[1]*s1 + scb[2]*s2 + sct[0];
        float t1 = scb[3]*s0 + scb[4]*s1 + scb[5]*s2 + sct[1];
        float t2 = scb[6]*s0 + scb[7]*s1 + scb[8]*s2 + sct[2];
        int gx = (int)((t0 + 54.0f) / 0.3f);    // trunc toward zero (matches ref)
        int gy = (int)((t1 + 54.0f) / 0.3f);
        int gz = (int)((t2 + 10.0f) / 20.0f);
        bool kept = (gx >= 0) && (gx < NXV) && (gy >= 0) && (gy < NYV) && (gz == 0);
        sc[tid] = kept ? (gx * NYV + gy) : -1;
    }
    __syncthreads();

    for (int i = tid; i < DBINS * OUTC; i += 128) {
        int d = i / OUTC, c = i - d * OUTC;
        int cell = sc[d];
        if (cell >= 0)
            atomicAdd(&bev[(size_t)c * BEVHW + cell], sw[d] * sf[c]);
    }
}

// ---------------------------------------------------------------------------
// Host launcher
// ---------------------------------------------------------------------------
extern "C" void kernel_launch(void* const* d_in, const int* in_sizes, int n_in,
                              void* d_out, int out_size, void* d_ws, size_t ws_size,
                              hipStream_t stream) {
    (void)in_sizes; (void)n_in; (void)out_size; (void)ws_size;

    const float* img      = (const float*)d_in[0];
    const float* depth    = (const float*)d_in[1];
    const float* nbd      = (const float*)d_in[2];
    const float* c2l_rot  = (const float*)d_in[3];
    const float* c2l_tr   = (const float*)d_in[4];
    const float* intr     = (const float*)d_in[5];
    const float* post_rot = (const float*)d_in[6];
    const float* post_tr  = (const float*)d_in[7];
    // params (dict insertion order)
    const float* attn_w = (const float*)d_in[8];
    const float* attn_b = (const float*)d_in[9];
    const float* d1_w   = (const float*)d_in[10];
    const float* d1_b   = (const float*)d_in[11];
    const float* bn1_g  = (const float*)d_in[12];
    const float* bn1_b  = (const float*)d_in[13];
    const float* d2_w   = (const float*)d_in[14];
    const float* d2_b   = (const float*)d_in[15];
    const float* bn2_g  = (const float*)d_in[16];
    const float* bn2_b  = (const float*)d_in[17];
    const float* d3_w   = (const float*)d_in[18];
    const float* d3_b   = (const float*)d_in[19];
    const float* bn3_g  = (const float*)d_in[20];
    const float* bn3_b  = (const float*)d_in[21];
    const float* p1_w   = (const float*)d_in[22];
    const float* p1_b   = (const float*)d_in[23];
    const float* bnp1_g = (const float*)d_in[24];
    const float* bnp1_b = (const float*)d_in[25];
    const float* p2_w   = (const float*)d_in[26];
    const float* p2_b   = (const float*)d_in[27];
    const float* bnp2_g = (const float*)d_in[28];
    const float* bnp2_b = (const float*)d_in[29];
    const float* p3_w   = (const float*)d_in[30];
    const float* p3_b   = (const float*)d_in[31];
    const float* ds1_w  = (const float*)d_in[32];
    const float* bnd1_g = (const float*)d_in[33];
    const float* bnd1_b = (const float*)d_in[34];
    const float* ds2_w  = (const float*)d_in[35];
    const float* bnd2_g = (const float*)d_in[36];
    const float* bnd2_b = (const float*)d_in[37];
    const float* ds3_w  = (const float*)d_in[38];
    const float* bnd3_g = (const float*)d_in[39];
    const float* bnd3_b = (const float*)d_in[40];

    float* ws = (float*)d_ws;
    // --- stage-1 region (aliased by stage-2 after p3) ---
    float* h1  = ws + 0;          // 6*8*256*704   = 8,650,752
    float* h2  = ws + 8650752;    // 6*32*64*176   = 2,162,688
    float* zc  = ws + 10813440;   // 6*320*2816    = 5,406,720
    float* p1o = ws + 16220160;   // 6*256*2816    = 4,325,376
    float* p2o = ws + 20545536;   // 6*256*2816    = 4,325,376  (end 24,870,912)
    // --- stage-2 aliases ---
    float* bev = ws + 0;          // 80*360*360    = 10,368,000
    float* y1  = ws + 10368000;   // 80*360*360    = 10,368,000
    float* y2  = ws + 20736000;   // 80*180*180    =  2,592,000 (end 23,328,000)
    // --- persistent ---
    float* zo  = ws + 24870912;   // 6*139*2816    = 2,348,544
    bhalf_t* wtb   = (bhalf_t*)(ws + 27219456);
    bhalf_t* wt_p1 = wtb;                 // 2880*256 = 737,280
    bhalf_t* wt_p2 = wt_p1 + 737280;      // 2304*256 = 589,824
    bhalf_t* wt_p3 = wt_p2 + 589824;      //  256*139 =  35,584
    bhalf_t* wt_d1 = wt_p3 + 35584;       //  720*80  =  57,600
    bhalf_t* wt_d2 = wt_d1 + 57600;
    bhalf_t* wt_d3 = wt_d2 + 57600;
    float* bb   = ws + 27987200;
    float* b_p1 = bb;        float* b_p2 = bb + 256;  float* b_p3 = bb + 512;
    float* b_d1 = bb + 672;  float* b_d2 = bb + 752;  float* b_d3 = bb + 832;

    // ---- weight prep (BN fold + bf16 [K][Cout] transpose) ----
    prep_weights_kernel<<<(2880*256+255)/256, 256, 0, stream>>>(p1_w, p1_b, bnp1_g, bnp1_b, wt_p1, b_p1, 256, 2880);
    prep_weights_kernel<<<(2304*256+255)/256, 256, 0, stream>>>(p2_w, p2_b, bnp2_g, bnp2_b, wt_p2, b_p2, 256, 2304);
    prep_weights_kernel<<<( 256*139+255)/256, 256, 0, stream>>>(p3_w, p3_b, nullptr, nullptr, wt_p3, b_p3, 139, 256);
    prep_weights_kernel<<<( 720* 80+255)/256, 256, 0, stream>>>(ds1_w, nullptr, bnd1_g, bnd1_b, wt_d1, b_d1, 80, 720);
    prep_weights_kernel<<<( 720* 80+255)/256, 256, 0, stream>>>(ds2_w, nullptr, bnd2_g, bnd2_b, wt_d2, b_d2, 80, 720);
    prep_weights_kernel<<<( 720* 80+255)/256, 256, 0, stream>>>(ds3_w, nullptr, bnd3_g, bnd3_b, wt_d3, b_d3, 80, 720);

    // ---- stage 1: attention fuse + dtransform ----
    attn_d1_kernel<<<(MIMG*IHH*IWW+255)/256, 256, 0, stream>>>(
        depth, nbd, attn_w, attn_b, d1_w, d1_b, bn1_g, bn1_b, h1);
    d2_kernel<<<(MIMG*32*64*176+255)/256, 256, 0, stream>>>(h1, d2_w, d2_b, bn2_g, bn2_b, h2);
    d3_kernel<<<(MIMG*64*NPIX+255)/256, 256, 0, stream>>>(h2, d3_w, d3_b, bn3_g, bn3_b, zc);
    img_concat_kernel<<<(MIMG*INCH*NPIX+255)/256, 256, 0, stream>>>(img, zc);

    // ---- depthnet (WMMA implicit GEMM) ----
    conv_wmma_kernel<3,1,1,1,FWW><<<dim3(22, 4, 6), 256, 0, stream>>>(
        zc,  wt_p1, b_p1, p1o, 320, 256, FHH, FWW, FHH);
    conv_wmma_kernel<3,1,1,1,FWW><<<dim3(22, 4, 6), 256, 0, stream>>>(
        p1o, wt_p2, b_p2, p2o, 256, 256, FHH, FWW, FHH);
    conv_wmma_kernel<1,1,0,0,FWW><<<dim3(22, 3, 6), 256, 0, stream>>>(
        p2o, wt_p3, b_p3, zo,  256, 139, FHH, FWW, FHH);

    // ---- BEV pool: zero grid, then softmax+geometry+atomic scatter ----
    zero_kernel<<<(OUTC*BEVHW+255)/256, 256, 0, stream>>>(bev, OUTC*BEVHW);
    bev_scatter_kernel<<<MIMG*NPIX, 128, 0, stream>>>(
        zo, c2l_rot, c2l_tr, intr, post_rot, post_tr, bev);

    // ---- downsample (WMMA implicit GEMM) ----
    conv_wmma_kernel<3,1,1,1,360><<<dim3((360*360+127)/128, 2, 1), 256, 0, stream>>>(
        bev, wt_d1, b_d1, y1, 80, 80, 360, 360, 360);
    conv_wmma_kernel<3,2,1,1,180><<<dim3((180*180+127)/128, 2, 1), 256, 0, stream>>>(
        y1,  wt_d2, b_d2, y2, 80, 80, 360, 360, 180);
    conv_wmma_kernel<3,1,1,1,180><<<dim3((180*180+127)/128, 2, 1), 256, 0, stream>>>(
        y2,  wt_d3, b_d3, (float*)d_out, 80, 80, 180, 180, 180);
}
